// GraphAttentionLayer_24524263260206
// MI455X (gfx1250) — matrix operand order repaired and checked by reading, not compiled, not run on hardware
//
#include <hip/hip_runtime.h>

#define F_IN 128
#define D_OUT 64
#define BN_EPS 1e-3f
#define SLOPE 0.2f

typedef __attribute__((ext_vector_type(16))) __bf16    v16bf;
typedef __attribute__((ext_vector_type(8)))  float     v8f;
typedef __attribute__((ext_vector_type(8)))  unsigned  v8u;

// bf16 <-> f32 via bit manipulation (round-to-nearest-even), storage kept as u16/u32
static __device__ __forceinline__ unsigned short f2bfu(float f) {
    unsigned u = __builtin_bit_cast(unsigned, f);
    return (unsigned short)((u + 0x7FFFu + ((u >> 16) & 1u)) >> 16);
}
static __device__ __forceinline__ float bfu2f(unsigned short h) {
    return __builtin_bit_cast(float, (unsigned)h << 16);
}

// ---------------- utility kernels ----------------

__global__ void fill_zero(float* p, long long count) {
    long long i = (long long)blockIdx.x * blockDim.x + threadIdx.x;
    long long st = (long long)gridDim.x * blockDim.x;
    for (; i < count; i += st) p[i] = 0.0f;
}

__global__ void tanh_inplace(float* p, long long count) {
    long long i = (long long)blockIdx.x * blockDim.x + threadIdx.x;
    long long st = (long long)gridDim.x * blockDim.x;
    for (; i < count; i += st) p[i] = tanhf(p[i]);
}

// Pre-swizzle a KxN f32 matrix (K=128 rows, N=ldn cols) into WMMA-B fragment order:
//   out[((nt*4 + kb)*32 + lane)*8 + j] = pack_bf16(B[k][N], B[k+1][N])
//   with N = nt*16 + (lane&15), k = kb*32 + ((j&4)?16:0) + (lane>>4)*8 + 2*(j&3)
// Hot loop then reads each fragment as 8 consecutive dwords per lane (2x b128).
__global__ void pack_bfrag(const float* __restrict__ B, unsigned* __restrict__ out, int ldn) {
    int ntiles = ldn >> 4;
    int total  = ntiles * 4 * 32 * 8;
    int idx = blockIdx.x * blockDim.x + threadIdx.x;
    int st  = gridDim.x * blockDim.x;
    for (; idx < total; idx += st) {
        int j    = idx & 7;
        int lane = (idx >> 3) & 31;
        int kb   = (idx >> 8) & 3;
        int nt   = idx >> 10;
        int N    = nt * 16 + (lane & 15);
        int k    = kb * 32 + ((j & 4) ? 16 : 0) + (lane >> 4) * 8 + 2 * (j & 3);
        unsigned lo = f2bfu(B[k * ldn + N]);
        unsigned hi = f2bfu(B[(k + 1) * ldn + N]);
        out[idx] = lo | (hi << 16);
    }
}

static __device__ __forceinline__ v16bf load_frag(const unsigned* __restrict__ fr,
                                                  int nt, int kb, int lane) {
    v8u d = *(const v8u*)(fr + (((nt * 4 + kb) * 32 + lane) << 3));
    return __builtin_bit_cast(v16bf, d);
}

// ---------------- batchnorm stats ----------------

__global__ __launch_bounds__(256) void bn_stats(const float* __restrict__ x, float* __restrict__ sums, int n) {
    const int f  = threadIdx.x & (F_IN - 1);
    const int rr = threadIdx.x >> 7;  // 0 or 1: two rows per block iteration
    float s = 0.f, s2 = 0.f;
    for (int row = blockIdx.x * 2 + rr; row < n; row += gridDim.x * 2) {
        float v = x[(size_t)row * F_IN + f];
        s += v; s2 += v * v;
    }
    __shared__ float ls[256], ls2[256];
    ls[threadIdx.x] = s; ls2[threadIdx.x] = s2;
    __syncthreads();
    if (threadIdx.x < F_IN) {
        atomicAdd(&sums[f],        ls[threadIdx.x]  + ls[threadIdx.x + F_IN]);
        atomicAdd(&sums[F_IN + f], ls2[threadIdx.x] + ls2[threadIdx.x + F_IN]);
    }
}

__global__ void bn_finalize(const float* __restrict__ sums, float* __restrict__ meansc, int n) {
    int f = threadIdx.x;
    if (f < F_IN) {
        float inv_n = 1.0f / (float)n;
        float m   = sums[f] * inv_n;
        float var = sums[F_IN + f] * inv_n - m * m;
        meansc[f]        = m;
        meansc[F_IN + f] = rsqrtf(var + BN_EPS);
    }
}

// ---------------- fused normalize + 3 GEMMs + quadratic forms (WMMA) ----------------
// One wave32 per 16-node tile; 4 waves / block.
// A layout (16x32 bf16, ISA 7.12.2): lane M = lane&15, half = lane>>4,
//   VGPR j: K = kb*32 + (j&4 ? 16:0) + half*8 + 2*(j&3) (+elem)   (contiguous 16B runs in LDS)
// B fragments: pre-swizzled by pack_bfrag -> 2x b128 per fragment per lane.
// C/D layout (16x16 f32): lane N = lane&15, VGPR v: M = v + 8*half
//
// Tail handling: mapped/a1/a2 are allocated padded to a multiple of 16 rows, so
// ALL stores are unguarded (pad rows absorb the clamped-row junk, never read later).

__global__ __launch_bounds__(128) void gat_gemm(
    const float* __restrict__ x, const float* __restrict__ meansc,
    const unsigned* __restrict__ Wf, const unsigned* __restrict__ K1f,
    const unsigned* __restrict__ K2f,
    float* __restrict__ mapped, float* __restrict__ a1, float* __restrict__ a2, int n)
{
    __shared__ unsigned short xs[4][16 * 136];   // padded stride -> conflict-free fragment gathers
    const int lane = threadIdx.x & 31;
    const int wv   = threadIdx.x >> 5;
    const int r0   = (blockIdx.x * 4 + wv) * 16;

    // stage normalized tile into LDS as bf16 (branchless row clamp for the pad tail)
    for (int i = lane; i < 16 * F_IN; i += 32) {
        int m = i >> 7, f = i & (F_IN - 1);
        int row = r0 + m;
        row = row < n ? row : n - 1;
        float v = (x[(size_t)row * F_IN + f] - meansc[f]) * meansc[F_IN + f];
        xs[wv][m * 136 + f] = f2bfu(v);
    }
    __syncthreads();

    const int M = lane & 15, half = lane >> 4;

    // build A fragments for the 4 K-blocks (K = 0..127)
    v16bf afrag[4];
#pragma unroll
    for (int kb = 0; kb < 4; ++kb) {
#pragma unroll
        for (int j = 0; j < 8; ++j) {
            int k0 = kb * 32 + ((j & 4) ? 16 : 0) + half * 8 + 2 * (j & 3);
            afrag[kb][2 * j]     = __builtin_bit_cast(__bf16, xs[wv][M * 136 + k0]);
            afrag[kb][2 * j + 1] = __builtin_bit_cast(__bf16, xs[wv][M * 136 + k0 + 1]);
        }
    }

    // mapped = xn @ W   (128x64)
    float* mpbase = mapped + (size_t)(r0 + 8 * half) * D_OUT + M;
#pragma unroll
    for (int nt = 0; nt < D_OUT / 16; ++nt) {
        v8f acc = {};
#pragma unroll
        for (int kb = 0; kb < 4; ++kb) {
            v16bf b = load_frag(Wf, nt, kb, lane);
            acc = __builtin_amdgcn_wmma_f32_16x16x32_bf16(false, afrag[kb], false, b,
                                                          (short)0, acc, false, false);
        }
        float* mp = mpbase + nt * 16;
#pragma unroll
        for (int v = 0; v < 8; ++v) mp[(size_t)v * D_OUT] = acc[v];   // const-offset stores
    }

    // quadratic forms: a1[m] = tanh(sum_n (xn@K1)[m][n] * xn[m][n]); same for K2
    float ra1[8], ra2[8];
#pragma unroll
    for (int v = 0; v < 8; ++v) { ra1[v] = 0.f; ra2[v] = 0.f; }

#pragma unroll 2
    for (int nt = 0; nt < F_IN / 16; ++nt) {
        v8f acc1 = {}, acc2 = {};
#pragma unroll
        for (int kb = 0; kb < 4; ++kb) {
            v16bf b1 = load_frag(K1f, nt, kb, lane);
            v16bf b2 = load_frag(K2f, nt, kb, lane);
            acc1 = __builtin_amdgcn_wmma_f32_16x16x32_bf16(false, afrag[kb], false, b1,
                                                           (short)0, acc1, false, false);
            acc2 = __builtin_amdgcn_wmma_f32_16x16x32_bf16(false, afrag[kb], false, b2,
                                                           (short)0, acc2, false, false);
        }
#pragma unroll
        for (int v = 0; v < 8; ++v) {
            float xv = bfu2f(xs[wv][(v + 8 * half) * 136 + nt * 16 + M]);
            ra1[v] += acc1[v] * xv;
            ra2[v] += acc2[v] * xv;
        }
    }

    // reduce across the 16 lanes (N dim) of each half
#pragma unroll
    for (int v = 0; v < 8; ++v) {
#pragma unroll
        for (int off = 8; off >= 1; off >>= 1) {
            ra1[v] += __shfl_xor(ra1[v], off, 16);
            ra2[v] += __shfl_xor(ra2[v], off, 16);
        }
    }
    if (M == 0) {
        int rb = r0 + 8 * half;
#pragma unroll
        for (int v = 0; v < 8; ++v) {
            a1[rb + v] = tanhf(ra1[v]);
            a2[rb + v] = tanhf(ra2[v]);
        }
    }
}

// ---------------- edge phase ----------------
// a1,a2 in (-1,1) => e = leaky_relu(a1+a2) in (-0.4, 2) => exp() safe without max-shift
// (softmax is shift-invariant, so result matches the reference's max-subtracted form)

__global__ __launch_bounds__(256) void edge_score(
    const int* __restrict__ src, const int* __restrict__ dst, const float* __restrict__ adj,
    const float* __restrict__ a1, const float* __restrict__ a2,
    float* __restrict__ ew, float* __restrict__ denom, int E)
{
    int e = blockIdx.x * blockDim.x + threadIdx.x;
    int st = gridDim.x * blockDim.x;
    for (; e < E; e += st) {
        int s = src[e], d = dst[e];
        float v = adj[e] * (a1[s] + a2[d]);
        float le = v > 0.0f ? v : SLOPE * v;
        float w = __expf(le);
        ew[e] = w;
        atomicAdd(&denom[s], w);
    }
}

// SpMM scatter: 64 threads (= D_OUT) per edge, 4 edges per 256-thread block iter.
__global__ __launch_bounds__(256) void edge_spmm(
    const int* __restrict__ src, const int* __restrict__ dst,
    const float* __restrict__ ew, const float* __restrict__ denom,
    const float* __restrict__ mapped, float* __restrict__ out, int E)
{
    const int t  = threadIdx.x & (D_OUT - 1);
    const int eo = threadIdx.x >> 6;
    for (int e = blockIdx.x * 4 + eo; e < E; e += gridDim.x * 4) {
        int s = src[e], d = dst[e];
        float attn = ew[e] / denom[s];
        atomicAdd(&out[(size_t)s * D_OUT + t], attn * mapped[(size_t)d * D_OUT + t]);
    }
}

// ---------------- launch ----------------

extern "C" void kernel_launch(void* const* d_in, const int* in_sizes, int n_in,
                              void* d_out, int out_size, void* d_ws, size_t ws_size,
                              hipStream_t stream) {
    const float* x   = (const float*)d_in[0];
    const int*   src = (const int*)d_in[1];
    const int*   dst = (const int*)d_in[2];
    const float* adj = (const float*)d_in[3];
    const float* W   = (const float*)d_in[4];
    const float* K1  = (const float*)d_in[5];
    const float* K2  = (const float*)d_in[6];
    const int n = in_sizes[0] / F_IN;
    const int E = in_sizes[1];
    const int tiles = (n + 15) / 16;
    const int nPad  = tiles * 16;            // padded row count for unguarded tile stores
    float* out = (float*)d_out;

    char* ws = (char*)d_ws;
    size_t off = 0;
    auto take = [&](size_t bytes) {
        char* p = ws + off;
        off = (off + bytes + 255) & ~(size_t)255;
        return p;
    };
    float*    stats  = (float*)take(2 * F_IN * sizeof(float));
    float*    meansc = (float*)take(2 * F_IN * sizeof(float));
    unsigned* Wf     = (unsigned*)take((size_t)(D_OUT / 16) * 4 * 32 * 8 * 4);  // 4096 dwords
    unsigned* K1f    = (unsigned*)take((size_t)(F_IN / 16) * 4 * 32 * 8 * 4);   // 8192 dwords
    unsigned* K2f    = (unsigned*)take((size_t)(F_IN / 16) * 4 * 32 * 8 * 4);
    float*    a1     = (float*)take((size_t)nPad * sizeof(float));
    float*    a2     = (float*)take((size_t)nPad * sizeof(float));
    float*    denom  = (float*)take((size_t)n * sizeof(float));
    float*    ewv    = (float*)take((size_t)E * sizeof(float));
    float*    mapped = (float*)take((size_t)nPad * D_OUT * sizeof(float));
    (void)ws_size; (void)n_in; (void)out_size;

    // BN stats
    fill_zero<<<1, 256, 0, stream>>>(stats, 2 * F_IN);
    bn_stats<<<512, 256, 0, stream>>>(x, stats, n);
    bn_finalize<<<1, 128, 0, stream>>>(stats, meansc, n);

    // weights -> bf16, pre-swizzled into WMMA B-fragment order (L2-resident)
    pack_bfrag<<<16, 256, 0, stream>>>(W,  Wf,  D_OUT);
    pack_bfrag<<<32, 256, 0, stream>>>(K1, K1f, F_IN);
    pack_bfrag<<<32, 256, 0, stream>>>(K2, K2f, F_IN);

    // fused normalize + WMMA GEMMs + row scalars
    gat_gemm<<<(tiles + 3) / 4, 128, 0, stream>>>(x, meansc, Wf, K1f, K2f, mapped, a1, a2, n);

    // edge softmax + SpMM
    fill_zero<<<1024, 256, 0, stream>>>(denom, n);
    fill_zero<<<4096, 256, 0, stream>>>(out, (long long)n * D_OUT);
    edge_score<<<(E + 255) / 256, 256, 0, stream>>>(src, dst, adj, a1, a2, ewv, denom, E);
    edge_spmm<<<16384, 256, 0, stream>>>(src, dst, ewv, denom, mapped, out, E);
    tanh_inplace<<<4096, 256, 0, stream>>>(out, (long long)n * D_OUT);
}
